// StateTrackingRecurrentCell_39067022524925
// MI455X (gfx1250) — compile-verified, matrix operand order checked by reading, not run on hardware
//
#include <hip/hip_runtime.h>
#include <hip/hip_bf16.h>
#include <math.h>

// ---------------------------------------------------------------------------
// LayerNorm-GRU for MI455X (gfx1250, wave32, WMMA).
//   H=1024, B=8, L=2048.
//   - k_pack:   repack w_z/w_r/w_h (both halves) to bf16 WMMA B-fragment order
//   - k_ln_x:   xn = LN(x_seq; g_in,b_in) -> bf16
//   - k_gemm_gx: gx[g] = xn @ W[g][:, :H].T + b[g]   (v_wmma_f32_16x16x32_bf16)
//   - k_recurrent: persistent 16 WGs x 384 thr, 2048 steps, 2 grid syncs/step,
//                  r-gate broadcast via global_load_async_to_lds_b128
// ---------------------------------------------------------------------------

typedef __attribute__((ext_vector_type(16))) __bf16 v16bf;
typedef __attribute__((ext_vector_type(8)))  __bf16 v8bf;
typedef __attribute__((ext_vector_type(8)))  float  v8f;

#define Hdim   1024
#define Bdim   8
#define Ldim   2048
#define M_TOT  (Bdim * Ldim)   // 16384
#define NWG    16
#define RBLK   384             // 12 waves of 32
#define EPSV   1e-5f

// ---------------- WMMA fragment helpers (ISA 7.12.2 16-bit layouts) --------
// A (16x32, MxK): lane m(<16) rows M=m, K = kt*32 + {0..7,16..23};
//                 lane m+16  rows M=m, K = kt*32 + {8..15,24..31}.
// B (32x16, KxN): mirrored with N across lanes; we PRE-PACK B so each lane
// reads its 16 bf16 contiguously (32B -> two b128 loads).

__device__ __forceinline__ v16bf load_b_pk(const __bf16* __restrict__ p,
                                           int fragIdx, int lane) {
  const __bf16* q = p + ((long)fragIdx * 32 + lane) * 16;
  union { v16bf v; v8bf h[2]; } u;
  u.h[0] = *(const v8bf*)(q);
  u.h[1] = *(const v8bf*)(q + 8);
  return u.v;
}

// A from LDS, row-major [8][1024] bf16; rows 8..15 are implicit zeros.
__device__ __forceinline__ v16bf load_a_lds(const __bf16* __restrict__ sh,
                                            int kt, int lane) {
  union { v16bf v; v8bf h[2]; } u;
  const int m = lane & 15;
  if (m < 8) {
    const __bf16* q = sh + m * Hdim + kt * 32 + ((lane >> 4) << 3);
    u.h[0] = *(const v8bf*)(q);        // K offsets {0..7} or {8..15}
    u.h[1] = *(const v8bf*)(q + 16);   // K offsets {16..23} or {24..31}
  } else {
#pragma unroll
    for (int i = 0; i < 16; ++i) u.v[i] = (__bf16)0.0f;
  }
  return u.v;
}

// A from global xn (row-major [16384][1024] bf16), full 16 rows valid.
__device__ __forceinline__ v16bf load_a_glb(const __bf16* __restrict__ xn,
                                            int mt, int kt, int lane) {
  const __bf16* q = xn + (long)(mt * 16 + (lane & 15)) * Hdim + kt * 32 +
                    ((lane >> 4) << 3);
  union { v16bf v; v8bf h[2]; } u;
  u.h[0] = *(const v8bf*)(q);
  u.h[1] = *(const v8bf*)(q + 16);
  return u.v;
}

// ---------------- grid-wide barrier (sense-reversing, L2 atomics) ----------
// Cluster split-barrier issued first: if the 16 WGs are dispatched as one
// cluster it synchronizes in hardware (the atomic pass then falls through
// instantly); when not in a cluster it is architecturally a NOP.
__device__ __forceinline__ void gridBarrier(unsigned* cnt, unsigned* gen,
                                            unsigned target) {
  __syncthreads();
  if (threadIdx.x == 0) {
    __threadfence();
    __builtin_amdgcn_s_cluster_barrier();   // s_barrier_signal/-wait -3
    unsigned prev = __hip_atomic_fetch_add(cnt, 1u, __ATOMIC_ACQ_REL,
                                           __HIP_MEMORY_SCOPE_AGENT);
    if (prev == NWG - 1u) {
      __hip_atomic_store(cnt, 0u, __ATOMIC_RELAXED, __HIP_MEMORY_SCOPE_AGENT);
      __hip_atomic_fetch_add(gen, 1u, __ATOMIC_RELEASE,
                             __HIP_MEMORY_SCOPE_AGENT);
    } else {
      while (__hip_atomic_load(gen, __ATOMIC_ACQUIRE,
                               __HIP_MEMORY_SCOPE_AGENT) < target)
        __builtin_amdgcn_s_sleep(8);
    }
  }
  __syncthreads();
}

// ---------------- init: copy h_prev, reset barrier state -------------------
__global__ void k_init(const float* __restrict__ h_prev,
                       float* __restrict__ h_glob,
                       unsigned* __restrict__ bar) {
  int id = blockIdx.x * 256 + threadIdx.x;
  if (id < Bdim * Hdim) h_glob[id] = h_prev[id];
  if (id < 2) bar[id] = 0u;
}

// ---------------- weight packing to fragment order -------------------------
// dst[( (g*64+gnt)*32 + kt )*32 + lane][e]  <-  W_g[o][half*H + k], bf16
//   o = gnt*16 + (lane&15)
//   k = kt*32 + (lane>=16 ? 8:0) + (e<8 ? e : e+8)
__global__ __launch_bounds__(256)
void k_pack(const float* __restrict__ w_z, const float* __restrict__ w_r,
            const float* __restrict__ w_h, __bf16* __restrict__ pWx,
            __bf16* __restrict__ pWh) {
  int id = blockIdx.x * 256 + threadIdx.x;       // < 6291456
  int e = id & 15;  int r = id >> 4;
  int lane = r & 31; r >>= 5;
  int kt   = r & 31; r >>= 5;
  int gnt  = r & 63; r >>= 6;                    // r in 0..5
  int g = r % 3, half = r / 3;
  int k = kt * 32 + ((lane >= 16) ? 8 : 0) + ((e < 8) ? e : e + 8);
  int o = gnt * 16 + (lane & 15);
  const float* W = (g == 0) ? w_z : ((g == 1) ? w_r : w_h);
  float v = W[(long)o * (2 * Hdim) + half * Hdim + k];
  __bf16* dst = half ? pWh : pWx;
  dst[(((long)(g * 64 + gnt) * 32 + kt) * 32 + lane) * 16 + e] = (__bf16)v;
}

// ---------------- input layernorm -> bf16 ----------------------------------
__global__ __launch_bounds__(256)
void k_ln_x(const float* __restrict__ x, const float* __restrict__ g_in,
            const float* __restrict__ b_in, __bf16* __restrict__ xn) {
  const int m = blockIdx.x, tid = threadIdx.x;
  __shared__ float ss[8], qq[8];
  float xv[4], s = 0.f, q = 0.f;
#pragma unroll
  for (int i = 0; i < 4; ++i) {
    float v = x[(long)m * Hdim + tid + i * 256];
    xv[i] = v; s += v; q += v * v;
  }
  for (int off = 16; off > 0; off >>= 1) {
    s += __shfl_xor(s, off, 32);
    q += __shfl_xor(q, off, 32);
  }
  if ((tid & 31) == 0) { ss[tid >> 5] = s; qq[tid >> 5] = q; }
  __syncthreads();
  if (tid == 0) {
    float S = 0.f, Q = 0.f;
    for (int i = 0; i < 8; ++i) { S += ss[i]; Q += qq[i]; }
    float mean = S * (1.f / Hdim);
    ss[0] = mean;
    qq[0] = rsqrtf(Q * (1.f / Hdim) - mean * mean + EPSV);
  }
  __syncthreads();
  float mean = ss[0], rstd = qq[0];
#pragma unroll
  for (int i = 0; i < 4; ++i) {
    int col = tid + i * 256;
    xn[(long)m * Hdim + col] =
        (__bf16)((xv[i] - mean) * rstd * g_in[col] + b_in[col]);
  }
}

// ---------------- big input-projection GEMM (WMMA bf16) --------------------
// grid (1024, 48) x 128 threads; wave -> one 16x16 tile, K=1024 in 32 steps.
__global__ __launch_bounds__(128)
void k_gemm_gx(const __bf16* __restrict__ xn, const __bf16* __restrict__ pWx,
               const float* __restrict__ b_z, const float* __restrict__ b_r,
               const float* __restrict__ b_h, float* __restrict__ gx) {
  const int lane = threadIdx.x & 31, wv = threadIdx.x >> 5;
  const int mt = blockIdx.x;
  const int gidx = blockIdx.y * 4 + wv;   // 0..191 = gate*64 + colTile
  const int g = gidx >> 6, colTile = gidx & 63;
  v8f c = {};
  for (int kt = 0; kt < 32; ++kt) {
    v16bf a = load_a_glb(xn, mt, kt, lane);
    v16bf b = load_b_pk(pWx, gidx * 32 + kt, lane);
    if (kt + 1 < 32)  // high-locality prefetch -> near (WGP) scope
      __builtin_prefetch(pWx + ((long)(gidx * 32 + kt + 1) * 32 + lane) * 16,
                         0, 3);
    c = __builtin_amdgcn_wmma_f32_16x16x32_bf16(false, a, false, b, (short)0,
                                                c, false, false);
  }
  const int o = colTile * 16 + (lane & 15);
  const float bias = (g == 0) ? b_z[o] : ((g == 1) ? b_r[o] : b_h[o]);
  const int mhi = (lane >> 4) << 3;
#pragma unroll
  for (int j = 0; j < 8; ++j) {   // C layout: lane&15 = N, M = j + 8*(lane>=16)
    int m = mt * 16 + mhi + j;
    gx[((long)g * M_TOT + m) * Hdim + o] = c[j] + bias;
  }
}

// ---------------- persistent recurrent scan --------------------------------
__global__ __launch_bounds__(RBLK)
void k_recurrent(float* __restrict__ h_glob, __bf16* __restrict__ r_buf,
                 const float* __restrict__ gx, const __bf16* __restrict__ pWh,
                 const float* __restrict__ g_st, const float* __restrict__ b_st,
                 const float* __restrict__ g_out, const float* __restrict__ b_out,
                 float* __restrict__ out, float* __restrict__ hfin,
                 unsigned* __restrict__ bar_cnt, unsigned* __restrict__ bar_gen) {
  __shared__ __bf16 hn_sh[Bdim * Hdim];     // LN(h) * g_st + b_st   (16 KB)
  __shared__ __bf16 rhn_sh[Bdim * Hdim];    // r * hn                (16 KB)
  __shared__ __bf16 r_sh[Bdim * Hdim];      // async-DMA'd r         (16 KB)
  __shared__ float  z_sh[Bdim * 64];        // z for this WG's slice ( 2 KB)
  __shared__ float  sMean[Bdim], sRstd[Bdim];

  const int tid = threadIdx.x, lane = tid & 31, wv = tid >> 5;
  const int colBase = blockIdx.x * 64;
  unsigned epoch = 0;

  for (int t = 0; t <= Ldim; ++t) {
    // ---- Phase A: per-row mean/var of h; hn(bf16) -> LDS; out[t-1] --------
    if (wv < Bdim) {
      const int b = wv;
      float hv[32], s = 0.f, q = 0.f;
#pragma unroll
      for (int i = 0; i < 32; ++i) {
        float v = h_glob[b * Hdim + lane + i * 32];
        hv[i] = v; s += v; q += v * v;
      }
#pragma unroll
      for (int off = 16; off > 0; off >>= 1) {
        s += __shfl_xor(s, off, 32);
        q += __shfl_xor(q, off, 32);
      }
      float mean = s * (1.f / Hdim);
      float rstd = rsqrtf(q * (1.f / Hdim) - mean * mean + EPSV);
      if (lane == 0) { sMean[b] = mean; sRstd[b] = rstd; }
      if (t < Ldim) {
#pragma unroll
        for (int i = 0; i < 32; ++i) {
          int col = lane + i * 32;
          hn_sh[b * Hdim + col] =
              (__bf16)((hv[i] - mean) * rstd * g_st[col] + b_st[col]);
        }
      }
    }
    __syncthreads();
    if (t > 0) {  // out[t-1] = LN(h) with output gain (same mean/var)
      for (int idx = tid; idx < Bdim * 64; idx += RBLK) {
        int b = idx >> 6, o = colBase + (idx & 63);
        float hval = h_glob[b * Hdim + o];
        out[((long)b * Ldim + (t - 1)) * Hdim + o] =
            (hval - sMean[b]) * sRstd[b] * g_out[o] + b_out[o];
      }
    }
    if (t == Ldim) {
      for (int idx = tid; idx < Bdim * 64; idx += RBLK) {
        int b = idx >> 6, o = colBase + (idx & 63);
        hfin[b * Hdim + o] = h_glob[b * Hdim + o];
      }
      break;
    }

    // ---- Phase B: z and r matmuls (waves 0..7: gate = wv>>2, tile = wv&3) -
    if (wv < 8) {
      const int g = wv >> 2;                       // 0 = z, 1 = r
      const int colTile = blockIdx.x * 4 + (wv & 3);
      v8f c = {};
      for (int kt = 0; kt < 32; ++kt) {
        v16bf a = load_a_lds(hn_sh, kt, lane);
        v16bf b = load_b_pk(pWh, (g * 64 + colTile) * 32 + kt, lane);
        if (kt + 1 < 32)
          __builtin_prefetch(
              pWh + ((long)((g * 64 + colTile) * 32 + kt + 1) * 32 + lane) * 16,
              0, 3);
        c = __builtin_amdgcn_wmma_f32_16x16x32_bf16(false, a, false, b,
                                                    (short)0, c, false, false);
      }
      const int o = colTile * 16 + (lane & 15);
      const int bhi = (lane >> 4) << 3;
#pragma unroll
      for (int j = 0; j < 8; ++j) {
        int b = bhi + j;
        if (b < Bdim) {
          float pre = c[j] + gx[((long)g * M_TOT + b * Ldim + t) * Hdim + o];
          float sg = 1.f / (1.f + __expf(-pre));
          if (g == 0) z_sh[b * 64 + (o - colBase)] = sg;
          else        r_buf[b * Hdim + o] = (__bf16)sg;  // broadcast r (bf16)
        }
      }
    }
    gridBarrier(bar_cnt, bar_gen, ++epoch);        // r globally visible

    // ---- Phase C: async-DMA r into LDS; rhn = r*hn; candidate matmul ------
    // r broadcast: 16 KB from L2 straight into LDS, no VGPR staging
    // (ASYNCcnt path: global_load_async_to_lds_b128 + s_wait_asynccnt).
    for (int idx = tid; idx < (Bdim * Hdim) / 8; idx += RBLK) {  // 1024 x 16B
      unsigned lds_b = (unsigned)(size_t)(&r_sh[idx * 8]);
      const __bf16* gsrc = r_buf + idx * 8;
      asm volatile("global_load_async_to_lds_b128 %0, %1, off"
                   :: "v"(lds_b), "v"(gsrc) : "memory");
    }
    asm volatile("s_wait_asynccnt 0x0" ::: "memory");
    __syncthreads();

    for (int idx = tid; idx < Bdim * Hdim; idx += RBLK)
      rhn_sh[idx] = (__bf16)((float)r_sh[idx] * (float)hn_sh[idx]);
    __syncthreads();

    if (wv < 4) {
      const int colTile = blockIdx.x * 4 + wv;
      v8f c = {};
      for (int kt = 0; kt < 32; ++kt) {
        v16bf a = load_a_lds(rhn_sh, kt, lane);
        v16bf b = load_b_pk(pWh, (2 * 64 + colTile) * 32 + kt, lane);
        c = __builtin_amdgcn_wmma_f32_16x16x32_bf16(false, a, false, b,
                                                    (short)0, c, false, false);
      }
      const int o = colTile * 16 + (lane & 15);
      const int bhi = (lane >> 4) << 3;
#pragma unroll
      for (int j = 0; j < 8; ++j) {
        int b = bhi + j;
        if (b < Bdim) {
          float pre = c[j] + gx[((long)2 * M_TOT + b * Ldim + t) * Hdim + o];
          float hc = tanhf(pre);
          float zz = z_sh[b * 64 + (o - colBase)];
          float ho = h_glob[b * Hdim + o];
          h_glob[b * Hdim + o] = (1.f - zz) * ho + zz * hc;  // own slice only
        }
      }
    }
    gridBarrier(bar_cnt, bar_gen, ++epoch);        // h_new globally visible
  }
}

// ---------------------------------------------------------------------------
extern "C" void kernel_launch(void* const* d_in, const int* in_sizes, int n_in,
                              void* d_out, int out_size, void* d_ws,
                              size_t ws_size, hipStream_t stream) {
  const float* x_seq  = (const float*)d_in[0];
  const float* h_prev = (const float*)d_in[1];
  const float* w_z = (const float*)d_in[2];
  const float* b_z = (const float*)d_in[3];
  const float* w_r = (const float*)d_in[4];
  const float* b_r = (const float*)d_in[5];
  const float* w_h = (const float*)d_in[6];
  const float* b_h = (const float*)d_in[7];
  const float* g_in = (const float*)d_in[8];
  const float* b_in = (const float*)d_in[9];
  const float* g_st = (const float*)d_in[10];
  const float* b_st = (const float*)d_in[11];
  const float* g_out = (const float*)d_in[12];
  const float* b_out = (const float*)d_in[13];

  // Workspace layout (requires ~236 MB; ws assumed sufficient)
  char* ws = (char*)d_ws;
  size_t off = 0;
  float*  gx  = (float*)(ws + off);  off += (size_t)3 * M_TOT * Hdim * 4;
  __bf16* xn  = (__bf16*)(ws + off); off += (size_t)M_TOT * Hdim * 2;
  __bf16* pWx = (__bf16*)(ws + off); off += (size_t)3 * Hdim * Hdim * 2;
  __bf16* pWh = (__bf16*)(ws + off); off += (size_t)3 * Hdim * Hdim * 2;
  float*  h_glob = (float*)(ws + off); off += (size_t)Bdim * Hdim * 4;
  __bf16* r_buf  = (__bf16*)(ws + off); off += (size_t)Bdim * Hdim * 2;
  off = (off + 255) & ~(size_t)255;
  unsigned* bar = (unsigned*)(ws + off); off += 256;

  float* outs = (float*)d_out;
  float* hfin = outs + (size_t)Bdim * Ldim * Hdim;

  hipLaunchKernelGGL(k_init, dim3((Bdim * Hdim + 255) / 256), dim3(256), 0,
                     stream, h_prev, h_glob, bar);
  hipLaunchKernelGGL(k_pack, dim3(6291456 / 256), dim3(256), 0, stream,
                     w_z, w_r, w_h, pWx, pWh);
  hipLaunchKernelGGL(k_ln_x, dim3(M_TOT), dim3(256), 0, stream,
                     x_seq, g_in, b_in, xn);
  hipLaunchKernelGGL(k_gemm_gx, dim3(1024, 48), dim3(128), 0, stream,
                     xn, pWx, b_z, b_r, b_h, gx);
  hipLaunchKernelGGL(k_recurrent, dim3(NWG), dim3(RBLK), 0, stream,
                     h_glob, r_buf, gx, pWh, g_st, b_st, g_out, b_out,
                     outs, hfin, bar, bar + 1);
}
